// SimpleSNN_72413148611058
// MI455X (gfx1250) — compile-verified, matrix operand order checked by reading, not run on hardware
//
#include <hip/hip_runtime.h>

// ---------------------------------------------------------------------------
// SNN (norse LIFCell x2) on CDNA5/gfx1250.
//  - 16 batch rows per workgroup (2 waves); wave w owns hidden cols [128w,128w+128)
//  - x tile (16x784 f32) staged once per timestep into LDS via the Tensor Data
//    Mover (tensor_load_to_lds, TENSORcnt), double-buffered, prefetched 1 step
//    ahead. A-fragments come from LDS (ds_load_b128) + v_perm bf16 pack.
//  - GEMMs on v_wmma_f32_16x16x32_bf16, f32 accumulate. LIF1 state in LDS,
//    LIF2 state + spike sum in VGPRs. K-tail peeled (main k-loop guard-free).
// ---------------------------------------------------------------------------

typedef __attribute__((ext_vector_type(16))) __bf16        v16bf;
typedef __attribute__((ext_vector_type(8)))  float         v8f;
typedef __attribute__((ext_vector_type(8)))  unsigned int  v8u;
typedef __attribute__((ext_vector_type(4)))  unsigned int  v4u;
typedef __attribute__((ext_vector_type(8)))  int           v8i;
typedef __attribute__((ext_vector_type(4)))  int           v4i;

#define T_STEPS 16
#define BATCH   4096
#define N_IN    784
#define N_HID   256
#define N_OUT   10
#define K1_PAD  800                        // W1 K padded to multiple of 32
#define W1BF_ELEMS (N_HID * K1_PAD)
#define W2BF_ELEMS (16 * N_HID)

#define ROWS_PER_WG 16
#define XROW        784                    // f32 per LDS x row (contiguous)
#define V1_STRIDE   260                    // lanes16-31 land +32 banks away
#define Z1_STRIDE   264                    // 528B rows, 16B aligned

#if __has_builtin(__builtin_amdgcn_tensor_load_to_lds)
#define USE_TDM 1
#endif

__device__ __forceinline__ unsigned int pk2_bf16(float lo, float hi) {
  // one v_perm_b32: D = { hi[31:16], lo[31:16] }  (exact for spike 0/1)
  return __builtin_amdgcn_perm(__builtin_bit_cast(unsigned int, hi),
                               __builtin_bit_cast(unsigned int, lo),
                               0x07060302u);
}

__device__ __forceinline__ unsigned short f32_to_bf16_rne(float f) {
  unsigned int u = __builtin_bit_cast(unsigned int, f);
  unsigned int r = 0x7FFFu + ((u >> 16) & 1u);
  return (unsigned short)((u + r) >> 16);
}

// ---- weight prep: f32 -> bf16, padded/zero-filled for clean K tails -------
__global__ void prep_w1(const float* __restrict__ W1, unsigned short* __restrict__ W1bf) {
  int id = blockIdx.x * blockDim.x + threadIdx.x;
  if (id >= W1BF_ELEMS) return;
  int h = id / K1_PAD, k = id - h * K1_PAD;
  W1bf[id] = f32_to_bf16_rne((k < N_IN) ? W1[h * N_IN + k] : 0.0f);
}

__global__ void prep_w2(const float* __restrict__ W2, unsigned short* __restrict__ W2bf) {
  int id = blockIdx.x * blockDim.x + threadIdx.x;
  if (id >= W2BF_ELEMS) return;
  int h = id / N_HID, k = id - h * N_HID;
  W2bf[id] = f32_to_bf16_rne((h < N_OUT) ? W2[h * N_HID + k] : 0.0f);
}

// ---- TDM: one descriptor moves the whole 16x784 f32 tile into LDS ---------
__device__ __forceinline__ void stage_x_tile(const float* gsrc, float* ldst,
                                             int tid_in_wg) {
#ifdef USE_TDM
  (void)tid_in_wg;
  const unsigned long long ga = (unsigned long long)(uintptr_t)gsrc;
  const unsigned int laddr = (unsigned int)(uintptr_t)ldst;   // LDS byte offset
  v4u g0;
  g0[0] = 1u;                                           // count=1 (valid D#)
  g0[1] = laddr;                                        // lds_addr
  g0[2] = (unsigned int)ga;                             // global_addr[31:0]
  g0[3] = (unsigned int)((ga >> 32) & 0x1FFFFFFu)       // global_addr[56:32]
        | (2u << 30);                                   // type = 2 ("image")
  v8i g1;
  g1[0] = (int)(2u << 16);                              // data_size = 4B
  g1[1] = (int)((unsigned)N_IN << 16);                  // tensor_dim0 = 784
  g1[2] = (int)(16u << 16);                             // tensor_dim1 = 16
  g1[3] = (int)((unsigned)N_IN << 16);                  // tile_dim0 = 784
  g1[4] = 16;                                           // tile_dim1=16, dim2=0
  g1[5] = N_IN;                                         // dim0_stride = 784
  g1[6] = 0;
  g1[7] = 0;
  const v4i zz4 = {0, 0, 0, 0};
  const v8i zz8 = {0, 0, 0, 0, 0, 0, 0, 0};
  __builtin_amdgcn_tensor_load_to_lds(g0, g1, zz4, zz4, zz8, 0);
#else
  // fallback copy (only wave 0 calls this path)
  const float4* s = (const float4*)gsrc;
  float4* d = (float4*)ldst;
  for (int i = tid_in_wg; i < ROWS_PER_WG * (XROW / 4); i += 64) d[i] = s[i];
#endif
}

__device__ __forceinline__ void stage_x_wait() {
#ifdef USE_TDM
  __builtin_amdgcn_s_wait_tensorcnt(0);
#endif
}

// ---- main persistent-state SNN kernel -------------------------------------
__global__ void __launch_bounds__(64, 4)
snn_kernel(const float* __restrict__ x,            // (T, B, 784) f32 spikes
           const float* __restrict__ b1,           // (256,)
           const float* __restrict__ b2,           // (10,)
           const unsigned short* __restrict__ W1bf,// (256, 800) bf16
           const unsigned short* __restrict__ W2bf,// (16, 256) bf16
           float* __restrict__ out)                // (B, 10) f32
{
  __shared__ __align__(16) float          xstage[2][ROWS_PER_WG * XROW]; // 100.4KB
  __shared__ __align__(16) float          s_v1[ROWS_PER_WG * V1_STRIDE]; // 16.6KB
  __shared__ __align__(16) float          s_i1[ROWS_PER_WG * V1_STRIDE]; // 16.6KB
  __shared__ __align__(16) unsigned short s_z1[ROWS_PER_WG * Z1_STRIDE]; //  8.4KB

  const int lane  = threadIdx.x & 31;
  const int wv    = threadIdx.x >> 5;          // wave 0/1 -> hidden half
  const int lhalf = (lane >> 4) & 1;
  const int lmod  = lane & 15;
  const int b0    = blockIdx.x * ROWS_PER_WG;
  const int nbase = wv * 128;                  // this wave's hidden column base

  // zero LIF-1 state cooperatively
  for (int idx = threadIdx.x; idx < ROWS_PER_WG * V1_STRIDE; idx += 64) {
    s_v1[idx] = 0.0f;
    s_i1[idx] = 0.0f;
  }

  // biases resident in VGPRs
  float b1v[8];
  #pragma unroll
  for (int nt = 0; nt < 8; ++nt) b1v[nt] = b1[nbase + nt * 16 + lmod];
  const float b2v = (lmod < N_OUT) ? b2[lmod] : 0.0f;

  // layer-2 LIF state + spike sum (C-fragment layout, both waves redundant)
  float v2s[8], i2s[8], zsum[8];
  #pragma unroll
  for (int j = 0; j < 8; ++j) { v2s[j] = 0.0f; i2s[j] = 0.0f; zsum[j] = 0.0f; }

  const float* xg = x + (size_t)b0 * N_IN;     // WG's rows in timestep 0

  // prefetch t=0 tile
  if (wv == 0) stage_x_tile(xg, xstage[0], threadIdx.x);

  for (int t = 0; t < T_STEPS; ++t) {
    if (wv == 0) stage_x_wait();
    __syncthreads();                            // x tile (t) visible to all

    // prefetch next timestep's tile into the other buffer
    if (wv == 0 && t + 1 < T_STEPS)
      stage_x_tile(xg + (size_t)(t + 1) * BATCH * N_IN, xstage[(t + 1) & 1],
                   threadIdx.x);

    const float* xb = xstage[t & 1] + lmod * XROW;   // this lane's A row

    // ---------------- layer 1 GEMM (this wave's 8 N-tiles) ----------------
    v8f acc[8];
    #pragma unroll
    for (int nt = 0; nt < 8; ++nt)
      #pragma unroll
      for (int j = 0; j < 8; ++j) acc[nt][j] = 0.0f;

    // 24 full K-chunks, completely guard-free
    for (int kc = 0; kc < 24; ++kc) {
      const int kb = kc * 32 + lhalf * 8;

      v8u afrag;
      {
        const float4* p0 = (const float4*)(xb + kb);
        const float4* p1 = (const float4*)(xb + kb + 16);
        const float4 f0 = p0[0], f1 = p0[1];
        const float4 f2 = p1[0], f3 = p1[1];
        afrag[0] = pk2_bf16(f0.x, f0.y);
        afrag[1] = pk2_bf16(f0.z, f0.w);
        afrag[2] = pk2_bf16(f1.x, f1.y);
        afrag[3] = pk2_bf16(f1.z, f1.w);
        afrag[4] = pk2_bf16(f2.x, f2.y);
        afrag[5] = pk2_bf16(f2.z, f2.w);
        afrag[6] = pk2_bf16(f3.x, f3.y);
        afrag[7] = pk2_bf16(f3.z, f3.w);
      }
      const v16bf a = __builtin_bit_cast(v16bf, afrag);

      #pragma unroll
      for (int nt = 0; nt < 8; ++nt) {
        const unsigned short* wrow =
            W1bf + (size_t)(nbase + nt * 16 + lmod) * K1_PAD + kb;
        const uint4 lo = *(const uint4*)(wrow);
        const uint4 hi = *(const uint4*)(wrow + 16);
        v8u bfr = { lo.x, lo.y, lo.z, lo.w, hi.x, hi.y, hi.z, hi.w };
        const v16bf bm = __builtin_bit_cast(v16bf, bfr);
        acc[nt] = __builtin_amdgcn_wmma_f32_16x16x32_bf16(
            false, a, false, bm, (short)0, acc[nt], false, false);
      }
    }

    // peeled K tail (kc=24): group1 (k in {768..783}) valid for ALL lanes,
    // group2 (k >= 784) zero for ALL lanes -> fully uniform, no divergence.
    {
      const int kb = 24 * 32 + lhalf * 8;          // 768 or 776
      v8u afrag;
      {
        const float4* p0 = (const float4*)(xb + kb);
        const float4 f0 = p0[0], f1 = p0[1];
        afrag[0] = pk2_bf16(f0.x, f0.y);
        afrag[1] = pk2_bf16(f0.z, f0.w);
        afrag[2] = pk2_bf16(f1.x, f1.y);
        afrag[3] = pk2_bf16(f1.z, f1.w);
        afrag[4] = 0u; afrag[5] = 0u; afrag[6] = 0u; afrag[7] = 0u;
      }
      const v16bf a = __builtin_bit_cast(v16bf, afrag);

      #pragma unroll
      for (int nt = 0; nt < 8; ++nt) {
        const unsigned short* wrow =
            W1bf + (size_t)(nbase + nt * 16 + lmod) * K1_PAD + kb;
        const uint4 lo = *(const uint4*)(wrow);
        const uint4 hi = *(const uint4*)(wrow + 16);  // in-bounds (K1_PAD=800)
        v8u bfr = { lo.x, lo.y, lo.z, lo.w, hi.x, hi.y, hi.z, hi.w };
        const v16bf bm = __builtin_bit_cast(v16bf, bfr);
        acc[nt] = __builtin_amdgcn_wmma_f32_16x16x32_bf16(
            false, a, false, bm, (short)0, acc[nt], false, false);
      }
    }

    // ---------------- LIF layer 1 on this wave's half, z1 -> LDS ----------
    #pragma unroll
    for (int nt = 0; nt < 8; ++nt) {
      const int n = nbase + nt * 16 + lmod;
      #pragma unroll
      for (int j = 0; j < 8; ++j) {
        const int m    = j + lhalf * 8;
        const int sidx = m * V1_STRIDE + n;
        const float vv = s_v1[sidx];
        const float ii = s_i1[sidx];
        const float vd = vv + 0.1f * (ii - vv);      // v + dt/tau_m*((0-v)+i)
        const float c1 = acc[nt][j] + b1v[nt];
        const bool  sp = vd > 1.0f;                  // heaviside(vd - v_th)
        s_v1[sidx] = sp ? 0.0f : vd;                 // hard reset
        s_i1[sidx] = 0.8f * ii + c1;                 // i*(1-dt/tau_s) + c1
        s_z1[m * Z1_STRIDE + n] = sp ? (unsigned short)0x3F80u
                                     : (unsigned short)0u;
      }
    }

    __syncthreads();                                 // z1 from both halves

    // ---------------- layer 2 GEMM (full K=256, both waves) ---------------
    v8f acc2;
    #pragma unroll
    for (int j = 0; j < 8; ++j) acc2[j] = 0.0f;

    #pragma unroll
    for (int kc = 0; kc < 8; ++kc) {
      const int kb = kc * 32 + lhalf * 8;
      const unsigned short* zp = s_z1 + lmod * Z1_STRIDE + kb;
      const uint4 zlo = *(const uint4*)(zp);
      const uint4 zhi = *(const uint4*)(zp + 16);
      v8u afr = { zlo.x, zlo.y, zlo.z, zlo.w, zhi.x, zhi.y, zhi.z, zhi.w };
      const v16bf a2 = __builtin_bit_cast(v16bf, afr);

      const unsigned short* wrow = W2bf + (size_t)lmod * N_HID + kb;
      const uint4 lo = *(const uint4*)(wrow);
      const uint4 hi = *(const uint4*)(wrow + 16);
      v8u bfr = { lo.x, lo.y, lo.z, lo.w, hi.x, hi.y, hi.z, hi.w };
      const v16bf bm = __builtin_bit_cast(v16bf, bfr);

      acc2 = __builtin_amdgcn_wmma_f32_16x16x32_bf16(
          false, a2, false, bm, (short)0, acc2, false, false);
    }

    // ---------------- LIF layer 2 + spike accumulation --------------------
    #pragma unroll
    for (int j = 0; j < 8; ++j) {
      const float c2 = acc2[j] + b2v;
      const float vd = v2s[j] + 0.1f * (i2s[j] - v2s[j]);
      const bool  sp = vd > 1.0f;
      v2s[j]  = sp ? 0.0f : vd;
      i2s[j]  = 0.8f * i2s[j] + c2;
      zsum[j] += sp ? 1.0f : 0.0f;
    }
  }

  // ---------------- write output (wave 0 only; both computed same) --------
  if (wv == 0 && lmod < N_OUT) {
    #pragma unroll
    for (int j = 0; j < 8; ++j) {
      const int m = j + lhalf * 8;
      out[(size_t)(b0 + m) * N_OUT + lmod] = zsum[j];
    }
  }
}

extern "C" void kernel_launch(void* const* d_in, const int* in_sizes, int n_in,
                              void* d_out, int out_size, void* d_ws, size_t ws_size,
                              hipStream_t stream) {
  const float* x  = (const float*)d_in[0];   // spike_input (T,B,784)
  const float* W1 = (const float*)d_in[1];   // (256,784)
  const float* b1 = (const float*)d_in[2];   // (256,)
  const float* W2 = (const float*)d_in[3];   // (10,256)
  const float* b2 = (const float*)d_in[4];   // (10,)
  float* out = (float*)d_out;                // (B,10)

  unsigned short* W1bf = (unsigned short*)d_ws;
  unsigned short* W2bf = W1bf + W1BF_ELEMS;

  prep_w1<<<(W1BF_ELEMS + 255) / 256, 256, 0, stream>>>(W1, W1bf);
  prep_w2<<<(W2BF_ELEMS + 255) / 256, 256, 0, stream>>>(W2, W2bf);

  const dim3 grid(BATCH / ROWS_PER_WG);      // 256 workgroups, 2 waves each
  snn_kernel<<<grid, 64, 0, stream>>>(x, b1, b2, W1bf, W2bf, out);
}
